// GaussianSampleST_37495064494574
// MI455X (gfx1250) — matrix operand order repaired and checked by reading, not compiled, not run on hardware
//
#include <hip/hip_runtime.h>
#include <hip/hip_bf16.h>

typedef __attribute__((ext_vector_type(16))) _Float16 v16h;
typedef __attribute__((ext_vector_type(8)))  _Float16 v8h;
typedef __attribute__((ext_vector_type(8)))  float    v8f;

#define N_B   16
#define C_CH  64
#define T_T   30
#define IN_H  64
#define IN_W  44
#define A_H   32
#define A_W   22
#define OUT_H 48
#define NT_TOT (N_B * T_T)
#define PAD_T 8

// ---------------------------------------------------------------------------
// Kernel 1: build normalized Gaussian filter banks.
//   FyG: [nt][32][64] f16 row-major (A-matrix for GEMM1)
//   FxG: [nt][32][64] f16 row-major, rows j=22..31 and cols w=44..63 zero
//        (row j = Fx row; used as B with K contiguous per row)
// ---------------------------------------------------------------------------
__global__ __launch_bounds__(64) void setup_filters(
    const float* __restrict__ dx, const float* __restrict__ dy,
    const float* __restrict__ ls2, const float* __restrict__ ld,
    const float* __restrict__ lg,
    _Float16* __restrict__ FyG, _Float16* __restrict__ FxG,
    float* __restrict__ gammaG)
{
    const int nt = blockIdx.x;
    const int n = nt / T_T, t = nt % T_T;
    const int r = threadIdx.x;

    const float* dxn = dx + n * T_T;
    const float* dyn = dy + n * T_T;
    // progressive offset, k=2: v[t] + 0.5*v[t-1] + 0.5*v[t-2]
    float pox = dxn[t] + 0.5f * ((t >= 1) ? dxn[t - 1] : 0.f)
                       + 0.5f * ((t >= 2) ? dxn[t - 2] : 0.f);
    float poy = dyn[t] + 0.5f * ((t >= 1) ? dyn[t - 1] : 0.f)
                       + 0.5f * ((t >= 2) ? dyn[t - 2] : 0.f);
    float dxv = tanhf(pox) * (IN_W * 0.5f) + (IN_W * 0.5f);   // *22 + 22
    float dyv = tanhf(poy) * (IN_H * 0.5f) + (IN_H * 0.5f);   // *32 + 32
    float sigma2 = expf(ls2[nt]);
    float delta  = expf(ld[nt]);
    float inv2s2 = 1.f / (2.f * sigma2);

    if (r == 0) gammaG[nt] = 1.f / (1.f + expf(-lg[nt]));

    _Float16* fy = FyG + (size_t)nt * (A_H * IN_H);   // [i*64 + h]
    _Float16* fx = FxG + (size_t)nt * (32 * 64);      // [j*64 + w], padded

    if (r < A_H) {
        // one Fy row i = r
        float mu = dyv + (r - A_H * 0.5f) * delta;
        float s = 0.f;
        for (int h = 0; h < IN_H; ++h) {
            float d = (float)h - mu;
            s += expf(-d * d * inv2s2);
        }
        float rn = 1.f / fmaxf(s, 1e-8f);
        for (int h = 0; h < IN_H; ++h) {
            float d = (float)h - mu;
            fy[r * IN_H + h] = (_Float16)(expf(-d * d * inv2s2) * rn);
        }
    } else if (r < A_H + A_W) {
        // one Fx row j = r - 32, row-major with zero pad w>=44
        int j = r - A_H;
        float mu = dxv + (j - A_W * 0.5f) * delta;
        float s = 0.f;
        for (int w = 0; w < IN_W; ++w) {
            float d = (float)w - mu;
            s += expf(-d * d * inv2s2);
        }
        float rn = 1.f / fmaxf(s, 1e-8f);
        for (int w = 0; w < 64; ++w) {
            float v = 0.f;
            if (w < IN_W) {
                float d = (float)w - mu;
                v = expf(-d * d * inv2s2) * rn;
            }
            fx[j * 64 + w] = (_Float16)v;
        }
    } else if (r < A_H + A_W + (32 - A_W)) {
        // zero pad rows j = 22..31
        int j = (r - (A_H + A_W)) + A_W;
        for (int w = 0; w < 64; ++w) fx[j * 64 + w] = (_Float16)0.f;
    }
}

// ---------------------------------------------------------------------------
// Kernel 2: temporal resampling weights
// ---------------------------------------------------------------------------
__global__ __launch_bounds__(512) void setup_temporal(
    const float* __restrict__ dt, const float* __restrict__ ldt,
    const float* __restrict__ lgt,
    float* __restrict__ tw0, float* __restrict__ tw1,
    int* __restrict__ ti0, int* __restrict__ ti1, float* __restrict__ gt)
{
    int idx = blockIdx.x * blockDim.x + threadIdx.x;
    if (idx < N_B) gt[idx] = 1.f / (1.f + expf(-lgt[idx]));
    if (idx >= NT_TOT) return;
    int n = idx / T_T, tq = idx % T_T;
    float dtv = tanhf(dt[n]) * (T_T * 0.5f) + (T_T * 0.5f);
    float delta_t = expf(ldt[n]);
    float mu = dtv + ((float)tq - T_T * 0.5f) * delta_t;
    mu = mu / (float)(T_T - 1) * 2.f - 1.f;
    float iy = ((mu + 1.f) * (float)T_T - 1.f) * 0.5f;
    float iy0 = floorf(iy);
    float w1 = iy - iy0, w0 = 1.f - w1;
    int i0 = (int)iy0, i1 = i0 + 1;
    float v0 = (i0 >= 0 && i0 < T_T) ? 1.f : 0.f;
    float v1 = (i1 >= 0 && i1 < T_T) ? 1.f : 0.f;
    ti0[idx] = min(max(i0, 0), T_T - 1);
    ti1[idx] = min(max(i1, 0), T_T - 1);
    tw0[idx] = w0 * v0;
    tw1[idx] = w1 * v1;
}

// ---------------------------------------------------------------------------
// helpers: build v16h WMMA fragments from two contiguous 16B LDS loads
// ---------------------------------------------------------------------------
__device__ __forceinline__ v16h frag_from2(const _Float16* p0, const _Float16* p1) {
    v8h lo = *(const v8h*)p0;
    v8h hi = *(const v8h*)p1;
    return __builtin_shufflevector(lo, hi, 0, 1, 2, 3, 4, 5, 6, 7,
                                           8, 9, 10, 11, 12, 13, 14, 15);
}

// ---------------------------------------------------------------------------
// Kernel 3: per-(n,t) bilinear glimpse via WMMA.
// G[c](32x22) = Fy(32x64) @ X_c(64x44) @ FxT(44x22), scaled by gamma.
// 8 waves/block; each wave owns 8 channels with private LDS X/Y slices.
//   sXt: X transposed [w(48)][h(64)]  -> B frags are contiguous 16B runs
//   sY : Y row-major  [i(32)][w(64)]  -> A frags are contiguous 16B runs
// ---------------------------------------------------------------------------
__global__ __launch_bounds__(256) void glimpse_gemm(
    const float* __restrict__ x,
    const _Float16* __restrict__ FyG, const _Float16* __restrict__ FxG,
    const float* __restrict__ gammaG, float* __restrict__ G)
{
    extern __shared__ __align__(16) char smem[];
    _Float16* sFy = (_Float16*)smem;            // 32*64
    _Float16* sFx = sFy + 32 * 64;              // 32*64 (padded rows/cols)
    _Float16* sXt = sFx + 32 * 64;              // 8 waves * 48*64
    _Float16* sY  = sXt + 8 * 48 * 64;          // 8 waves * 32*64

    const int nt    = blockIdx.x;
    const int tid   = threadIdx.x;
    const int wave  = tid >> 5;
    const int lane  = tid & 31;
    const int lmod  = lane & 15;
    const int lhalf = lane >> 4;            // 0 or 1
    const int khA   = lhalf * 8;            // A-frag K base within a 16-chunk
    const int kbB   = lhalf * 16;           // B-frag K base within a 32-chunk

    {   // cooperative fill of Fy / Fx
        const _Float16* s1 = FyG + (size_t)nt * 2048;
        const _Float16* s2 = FxG + (size_t)nt * 2048;
        for (int i = tid; i < 2048; i += 256) { sFy[i] = s1[i]; sFx[i] = s2[i]; }
    }
    const float gamma = gammaG[nt];
    __syncthreads();

    const int n = nt / T_T, t = nt % T_T;
    _Float16* myXt = sXt + wave * (48 * 64);
    _Float16* myY  = sY  + wave * (32 * 64);

    // zero pad rows of Xt (w=44..47) and all of Y once
    // (Y cols 48..63 must stay zero; 0..47 rewritten every channel)
    for (int i = lane; i < 4 * 64; i += 32) myXt[44 * 64 + i] = (_Float16)0.f;
    for (int i = lane; i < 32 * 64; i += 32) myY[i] = (_Float16)0.f;

    for (int cc = 0; cc < 8; ++cc) {
        const int c = wave + cc * 8;
        const float* xc = x + (((size_t)(n * C_CH + c) * T_T + t) * IN_H * IN_W);

        // stage X_c transposed as f16: myXt[w*64 + h], float2 global loads
        for (int i = lane; i < IN_H * (IN_W / 2); i += 32) {
            int h = i / (IN_W / 2);
            int w = (i - h * (IN_W / 2)) * 2;
            float2 v = *(const float2*)(xc + h * IN_W + w);
            myXt[w * 64 + h]       = (_Float16)v.x;
            myXt[(w + 1) * 64 + h] = (_Float16)v.y;
        }

        // prefetch next channel's tile while we compute (global_prefetch_b8)
        if (cc + 1 < 8) {
            const float* xn = x + (((size_t)(n * C_CH + (c + 8)) * T_T + t) * IN_H * IN_W);
            __builtin_prefetch(xn + lane * 32, 0, 3);
            __builtin_prefetch(xn + 1024 + lane * 32, 0, 3);
            __builtin_prefetch(xn + 2048 + lane * 32, 0, 3);
        }

        // ---- GEMM1: Y(32x48) = Fy(32x64) @ X(64x48) ----
        for (int mt = 0; mt < 2; ++mt) {
            const int row = mt * 16 + lmod;
            for (int ntile = 0; ntile < 3; ++ntile) {
                const int wcol = ntile * 16 + lmod;
                v8f acc = {};
#pragma unroll
                for (int ks = 0; ks < 2; ++ks) {
                    // A: lane holds row `row`, K = khA + {0..7, 16..23} + 32*ks
                    v16h a = frag_from2(&sFy[row * 64 + ks * 32 + khA],
                                        &sFy[row * 64 + ks * 32 + khA + 16]);
                    // B: lane holds col `wcol`, K = khA + {0..7, 16..23} + 32*ks
                    v16h b = frag_from2(&myXt[wcol * 64 + ks * 32 + khA],
                                        &myXt[wcol * 64 + ks * 32 + khA + 16]);
                    acc = __builtin_amdgcn_wmma_f32_16x16x32_f16(
                        false, a, false, b, (short)0, acc, false, false);
                }
#pragma unroll
                for (int v = 0; v < 8; ++v) {
                    int i = mt * 16 + v + 8 * lhalf;          // D: M = v + 8*lhalf
                    myY[i * 64 + wcol] = (_Float16)acc[v];
                }
            }
        }

        // ---- GEMM2: G_c(32x22) = Y(32x64) @ FxT(64x32) ----
        float* gout = G + ((size_t)nt * C_CH + c) * (A_H * A_W);
        for (int mt = 0; mt < 2; ++mt) {
            const int row = mt * 16 + lmod;
            for (int ntile = 0; ntile < 2; ++ntile) {
                const int jcol = ntile * 16 + lmod;
                v8f acc = {};
#pragma unroll
                for (int ks = 0; ks < 2; ++ks) {
                    v16h a = frag_from2(&myY[row * 64 + ks * 32 + khA],
                                        &myY[row * 64 + ks * 32 + khA + 16]);
                    v16h b = frag_from2(&sFx[jcol * 64 + ks * 32 + khA],
                                        &sFx[jcol * 64 + ks * 32 + khA + 16]);
                    acc = __builtin_amdgcn_wmma_f32_16x16x32_f16(
                        false, a, false, b, (short)0, acc, false, false);
                }
                if (jcol < A_W) {
#pragma unroll
                    for (int v = 0; v < 8; ++v) {
                        int i = mt * 16 + v + 8 * lhalf;
                        gout[i * A_W + jcol] = gamma * acc[v];
                    }
                }
            }
        }
    }
}

// ---------------------------------------------------------------------------
// Kernel 4: temporal 2-tap resample, gamma_t scale, pad 32 -> 48 rows.
// out shape (N, C, T, 48, 22)
// ---------------------------------------------------------------------------
__global__ __launch_bounds__(256) void resample_out(
    const float* __restrict__ G,
    const float* __restrict__ tw0, const float* __restrict__ tw1,
    const int* __restrict__ ti0, const int* __restrict__ ti1,
    const float* __restrict__ gt, float* __restrict__ out, int total)
{
    int idx = blockIdx.x * blockDim.x + threadIdx.x;
    if (idx >= total) return;
    int j  = idx % A_W;
    int y  = (idx / A_W) % OUT_H;
    int tq = (idx / (A_W * OUT_H)) % T_T;
    int c  = (idx / (A_W * OUT_H * T_T)) % C_CH;
    int n  = idx / (A_W * OUT_H * T_T * C_CH);
    float val = 0.f;
    if (y >= PAD_T && y < PAD_T + A_H) {
        int i = y - PAD_T;
        int nti = n * T_T + tq;
        size_t b0 = (((size_t)(n * T_T + ti0[nti]) * C_CH + c) * (A_H * A_W)) + i * A_W + j;
        size_t b1 = (((size_t)(n * T_T + ti1[nti]) * C_CH + c) * (A_H * A_W)) + i * A_W + j;
        val = 0.5f * gt[n] * (tw0[nti] * G[b0] + tw1[nti] * G[b1]);
    }
    out[idx] = val;
}

// ---------------------------------------------------------------------------
extern "C" void kernel_launch(void* const* d_in, const int* in_sizes, int n_in,
                              void* d_out, int out_size, void* d_ws, size_t ws_size,
                              hipStream_t stream) {
    const float* x   = (const float*)d_in[0];
    const float* dx  = (const float*)d_in[1];
    const float* dy  = (const float*)d_in[2];
    const float* ls2 = (const float*)d_in[3];
    const float* ld  = (const float*)d_in[4];
    const float* lg  = (const float*)d_in[5];
    const float* dt  = (const float*)d_in[6];
    const float* ldt = (const float*)d_in[7];
    const float* lgt = (const float*)d_in[8];
    float* out = (float*)d_out;

    char* ws = (char*)d_ws;
    size_t off = 0;
    auto alloc = [&](size_t bytes) -> void* {
        void* p = ws + off;
        off = (off + bytes + 255) & ~(size_t)255;
        return p;
    };
    float*    G    = (float*)alloc((size_t)NT_TOT * C_CH * A_H * A_W * sizeof(float));
    _Float16* FyG  = (_Float16*)alloc((size_t)NT_TOT * 2048 * sizeof(_Float16));
    _Float16* FxG  = (_Float16*)alloc((size_t)NT_TOT * 2048 * sizeof(_Float16));
    float*    gamG = (float*)alloc(NT_TOT * sizeof(float));
    float*    tw0  = (float*)alloc(NT_TOT * sizeof(float));
    float*    tw1  = (float*)alloc(NT_TOT * sizeof(float));
    int*      ti0  = (int*)alloc(NT_TOT * sizeof(int));
    int*      ti1  = (int*)alloc(NT_TOT * sizeof(int));
    float*    gt   = (float*)alloc(N_B * sizeof(float));

    setup_filters<<<NT_TOT, 64, 0, stream>>>(dx, dy, ls2, ld, lg, FyG, FxG, gamG);
    setup_temporal<<<1, 512, 0, stream>>>(dt, ldt, lgt, tw0, tw1, ti0, ti1, gt);

    size_t shmem = (size_t)(32 * 64 + 32 * 64 + 8 * 48 * 64 + 8 * 32 * 64) * sizeof(_Float16);
    glimpse_gemm<<<NT_TOT, 256, shmem, stream>>>(x, FyG, FxG, gamG, G);

    int total = N_B * C_CH * T_T * OUT_H * A_W;
    resample_out<<<(total + 255) / 256, 256, 0, stream>>>(G, tw0, tw1, ti0, ti1, gt, out, total);
}
